// GrokOmega_27230092657281
// MI455X (gfx1250) — matrix operand order repaired and checked by reading, not compiled, not run on hardware
//
#include <hip/hip_runtime.h>

// ============================================================================
// GrokOmega wave-evolution + interference decode for MI455X (gfx1250, wave32)
//
// Pipeline:
//   1) wave_init:  psi[4096x1024] fp32 (re/im) from char codes (sin/cos VALU)
//   2) 10 x { evolve (complex GEMM psi@H^T via V_WMMA_F32_16X16X4_F32, fp32),
//             normalize (row 2-norm) }
//   3) cvt: psi -> f16
//   4) decode: fused 5-way GEMM (M=4096,N=32000,K=1024) via
//      v_wmma_f32_16x16x32_f16:  |conj(psi)·pat|^2 + psi_re·dec_w^T + b
//      - A fragments loaded directly from L2-resident psi-f16 (two b128/lane,
//        in contiguous K order k = 16*(lane>>4) + s, matched by B)
//      - B tiles (patterns/dec_w, fp32->f16) double-buffered in LDS
//      - global_prefetch for the tile after next
// Float WMMA has no A/B negate modifier (NEG = CNeg only), so subtraction
// terms use an explicitly negated A fragment (folds to v_xor sign flip).
// ============================================================================

#define SEQ   4096
#define DIM   1024
#define VOCAB 32000
#define TWO_PI 6.28318530717958647692f

typedef __attribute__((ext_vector_type(16))) _Float16 v16h;
typedef __attribute__((ext_vector_type(8)))  _Float16 v8h;
typedef __attribute__((ext_vector_type(8)))  float    v8f;
typedef __attribute__((ext_vector_type(2)))  float    v2f;

#if defined(__has_builtin)
#if __has_builtin(__builtin_amdgcn_wmma_f32_16x16x4_f32)
#define HAVE_WMMA_F32 1
#endif
#endif
#ifndef HAVE_WMMA_F32
#define HAVE_WMMA_F32 0
#endif

// ---------------------------------------------------------------------------
// Kernel 1: Padilha wave + quaternion map -> psi (re, im=0), fp32
// ---------------------------------------------------------------------------
__global__ __launch_bounds__(256) void wave_init_kernel(
    const int* __restrict__ codes,
    float* __restrict__ pre, float* __restrict__ pim) {
  int gid = blockIdx.x * 256 + threadIdx.x;
  const int nblk = DIM / 4;
  int s = gid / nblk;
  int b = gid - s * nblk;
  if (s >= SEQ) return;

  float lam = (float)codes[s] * (1.0f / 256.0f);
  float t   = (float)s * (1.0f / (float)SEQ);
  float wt  = sinf(TWO_PI * t + 1.5f * lam);
  float ph0 = TWO_PI * t - TWO_PI * lam + 0.8f * lam * lam;

  int d0 = b * 4;
  float cr[4], si[4];
#pragma unroll
  for (int j = 0; j < 4; ++j) {
    float ph = ph0 + ((float)(d0 + j) * (1.0f / (float)DIM)) * TWO_PI;
    cr[j] = wt * cosf(ph);
    si[j] = wt * sinf(ph);
  }
  float* dr = pre + (size_t)s * DIM + d0;
  float* di = pim + (size_t)s * DIM + d0;
  dr[0] = cr[0];
  dr[1] = si[1];
  dr[2] = cr[2] * si[3];
  dr[3] = cr[3] * si[2];
  di[0] = 0.f; di[1] = 0.f; di[2] = 0.f; di[3] = 0.f;
}

// ---------------------------------------------------------------------------
// Kernel 2: one evolution step  q = p + (-i*dt/hbar)*(p @ H^T)  (unnormalized)
// Complex GEMM in fp32 via V_WMMA_F32_16X16X4_F32.
// ---------------------------------------------------------------------------
#define ET_K 16

__global__ __launch_bounds__(256) void evolve_kernel(
    const float* __restrict__ pre, const float* __restrict__ pim,
    const float* __restrict__ H,      // complex64 interleaved [DIM][DIM]
    const float* __restrict__ hbar,
    float* __restrict__ qre, float* __restrict__ qim) {
  __shared__ float sAre[64][ET_K];
  __shared__ float sAim[64][ET_K];
  __shared__ float sBre[64][ET_K];   // sB[n][k] = H[dbase+n][k0+k]
  __shared__ float sBim[64][ET_K];

  const int tid  = threadIdx.x;
  const int lane = tid & 31;
  const int wave = tid >> 5;
  const int sbase = blockIdx.y * 64;
  const int dbase = blockIdx.x * 64;
  const int m0 = (wave & 3) * 16;
  const int n0 = (wave >> 2) * 32;

  v8f z = {0.f,0.f,0.f,0.f,0.f,0.f,0.f,0.f};
  v8f accRe[2] = {z, z};
  v8f accIm[2] = {z, z};

  for (int k0 = 0; k0 < DIM; k0 += ET_K) {
    __syncthreads();
    for (int i = tid; i < 64 * ET_K; i += 256) {
      int r = i >> 4, c = i & 15;
      size_t ga = (size_t)(sbase + r) * DIM + k0 + c;
      sAre[r][c] = pre[ga];
      sAim[r][c] = pim[ga];
      const float* h2 = H + 2u * ((size_t)(dbase + r) * DIM + (k0 + c));
      sBre[r][c] = h2[0];
      sBim[r][c] = h2[1];
    }
    __syncthreads();

#if HAVE_WMMA_F32
#pragma unroll
    for (int kb = 0; kb < ET_K; kb += 4) {
      const int m  = m0 + (lane & 15);
      const int kk = kb + ((lane >> 4) << 1);
      v2f are; are[0] = sAre[m][kk]; are[1] = sAre[m][kk + 1];
      v2f aim; aim[0] = sAim[m][kk]; aim[1] = sAim[m][kk + 1];
      v2f aimn = -aim;   // no neg_a modifier on float WMMA
#pragma unroll
      for (int j = 0; j < 2; ++j) {
        const int n = n0 + j * 16 + (lane & 15);
        v2f bre; bre[0] = sBre[n][kk]; bre[1] = sBre[n][kk + 1];
        v2f bim; bim[0] = sBim[n][kk]; bim[1] = sBim[n][kk + 1];
        // Re(out) += a_re*b_re - a_im*b_im ;  Im(out) += a_re*b_im + a_im*b_re
        accRe[j] = __builtin_amdgcn_wmma_f32_16x16x4_f32(false, are,  false, bre,
                                                         (short)0, accRe[j], false, false);
        accRe[j] = __builtin_amdgcn_wmma_f32_16x16x4_f32(false, aimn, false, bim,
                                                         (short)0, accRe[j], false, false);
        accIm[j] = __builtin_amdgcn_wmma_f32_16x16x4_f32(false, are,  false, bim,
                                                         (short)0, accIm[j], false, false);
        accIm[j] = __builtin_amdgcn_wmma_f32_16x16x4_f32(false, aim,  false, bre,
                                                         (short)0, accIm[j], false, false);
      }
    }
#else
#pragma unroll
    for (int j = 0; j < 2; ++j) {
#pragma unroll
      for (int r = 0; r < 8; ++r) {
        int m = m0 + r + ((lane >> 4) << 3);
        int n = n0 + j * 16 + (lane & 15);
        float sre = 0.f, sim = 0.f;
        for (int kk = 0; kk < ET_K; ++kk) {
          float ar = sAre[m][kk], ai = sAim[m][kk];
          float br = sBre[n][kk], bi = sBim[n][kk];
          sre += ar * br - ai * bi;
          sim += ar * bi + ai * br;
        }
        accRe[j][r] += sre;
        accIm[j][r] += sim;
      }
    }
#endif
  }

  // q = p + (-i*dt/hbar)*out  => q_re = p_re + dt/h*Im, q_im = p_im - dt/h*Re
  const float dth = 0.1f / hbar[0];
#pragma unroll
  for (int j = 0; j < 2; ++j) {
#pragma unroll
    for (int r = 0; r < 8; ++r) {
      int srow = sbase + m0 + r + ((lane >> 4) << 3);
      int dcol = dbase + n0 + j * 16 + (lane & 15);
      size_t idx = (size_t)srow * DIM + dcol;
      qre[idx] = pre[idx] + dth * accIm[j][r];
      qim[idx] = pim[idx] - dth * accRe[j][r];
    }
  }
}

// ---------------------------------------------------------------------------
// Kernel 3: row normalize  out = q / (||q||_2 + 1e-8), one block per row
// ---------------------------------------------------------------------------
__global__ __launch_bounds__(256) void normalize_kernel(
    const float* __restrict__ qre, const float* __restrict__ qim,
    float* __restrict__ ore, float* __restrict__ oim) {
  __shared__ float red[256];
  const int s = blockIdx.x;
  const float* rr = qre + (size_t)s * DIM;
  const float* ri = qim + (size_t)s * DIM;
  float acc = 0.f;
  for (int d = threadIdx.x; d < DIM; d += 256) {
    float a = rr[d], b = ri[d];
    acc += a * a + b * b;
  }
  red[threadIdx.x] = acc;
  __syncthreads();
  for (int off = 128; off > 0; off >>= 1) {
    if (threadIdx.x < off) red[threadIdx.x] += red[threadIdx.x + off];
    __syncthreads();
  }
  const float scale = 1.0f / (sqrtf(red[0]) + 1e-8f);
  for (int d = threadIdx.x; d < DIM; d += 256) {
    ore[(size_t)s * DIM + d] = rr[d] * scale;
    oim[(size_t)s * DIM + d] = ri[d] * scale;
  }
}

// ---------------------------------------------------------------------------
// Kernel 4: fp32 -> f16 convert of psi for the decode GEMM
// ---------------------------------------------------------------------------
__global__ __launch_bounds__(256) void cvt_kernel(
    const float* __restrict__ re, const float* __restrict__ im,
    _Float16* __restrict__ hre, _Float16* __restrict__ him, int n) {
  int i = blockIdx.x * 256 + threadIdx.x;
  if (i < n) {
    hre[i] = (_Float16)re[i];
    him[i] = (_Float16)im[i];
  }
}

// ---------------------------------------------------------------------------
// Kernel 5: decode. out[s,v] = |conj(psi_s)·pat_v|^2 + psi_re_s·dec_w_v + b_v
// Fused 5-product, 3-accumulator GEMM on v_wmma_f32_16x16x32_f16.
// ---------------------------------------------------------------------------
#define DT_K   32
#define DT_NT  (DIM / DT_K)   // 32 K-slices

// A fragment (16-bit A 16x32): lane = M (&15); halves 0..15 = K hi..hi+15,
// hi = 16*(lane>>4)  -> two contiguous b128 loads.
__device__ __forceinline__ v16h load_a_direct(const _Float16* rowk0, int lane) {
  const int hi = (lane >> 4) << 4;
  const v8h lo = *(const v8h*)(rowk0 + hi);
  const v8h hq = *(const v8h*)(rowk0 + hi + 8);
  v16h a;
#pragma unroll
  for (int i = 0; i < 8; ++i) {
    a[i]     = lo[i];
    a[i + 8] = hq[i];
  }
  return a;
}

__global__ __launch_bounds__(256) void decode_kernel(
    const _Float16* __restrict__ pre, const _Float16* __restrict__ pim,
    const float* __restrict__ patterns,   // complex64 interleaved [VOCAB][DIM]
    const float* __restrict__ dec_w,
    const float* __restrict__ dec_b,
    float* __restrict__ out) {
  // double-buffered B tiles (fp32 -> f16 converted on the fly)
  __shared__ __attribute__((aligned(32))) _Float16 sBre[2][64][DT_K];
  __shared__ __attribute__((aligned(32))) _Float16 sBim[2][64][DT_K];
  __shared__ __attribute__((aligned(32))) _Float16 sW  [2][64][DT_K];

  const int tid  = threadIdx.x;
  const int lane = tid & 31;
  const int wave = tid >> 5;
  const int sbase = blockIdx.x * 64;   // x fastest => one V-tile's blocks co-run
  const int vbase = blockIdx.y * 64;
  const int m0 = (wave & 3) * 16;
  const int n0 = (wave >> 2) * 32;

  v8f z = {0.f,0.f,0.f,0.f,0.f,0.f,0.f,0.f};
  v8f aRe[2] = {z, z};
  v8f aIm[2] = {z, z};
  v8f aLn[2] = {z, z};

  auto load_btile = [&](int k0, int buf) {
    for (int i = tid; i < 64 * DT_K; i += 256) {
      int r = i >> 5, c = i & 31;
      size_t gb = (size_t)(vbase + r) * DIM + k0 + c;
      const float* p2 = patterns + 2u * gb;
      sBre[buf][r][c] = (_Float16)p2[0];
      sBim[buf][r][c] = (_Float16)p2[1];
      sW[buf][r][c]   = (_Float16)dec_w[gb];
    }
  };

  // prime the pipeline
  load_btile(0, 0);
  __syncthreads();

  // A row base pointers for this lane (psi f16 is L2-resident, reused 500x)
  const _Float16* arow_re = pre + (size_t)(sbase + m0 + (lane & 15)) * DIM;
  const _Float16* arow_im = pim + (size_t)(sbase + m0 + (lane & 15)) * DIM;
  const int kh = (lane >> 4) << 4;   // B frag: lanes 0-15 -> K 0..15, 16-31 -> 16..31

  for (int t = 0; t < DT_NT; ++t) {
    const int cur = t & 1;
    const int k0  = t * DT_K;

    // prefetch the tile after next (streams patterns/dec_w once from HBM)
    if (t + 2 < DT_NT) {
      int r = tid >> 2, c = (tid & 3) * 8;
      size_t gp = (size_t)(vbase + r) * DIM + (t + 2) * DT_K + c;
      __builtin_prefetch(patterns + 2u * gp, 0, 1);
      __builtin_prefetch(dec_w + gp, 0, 1);
    }
    // stage next tile into the other buffer while computing on this one
    if (t + 1 < DT_NT) load_btile(k0 + DT_K, cur ^ 1);

    v16h are  = load_a_direct(arow_re + k0, lane);
    v16h aim  = load_a_direct(arow_im + k0, lane);
    v16h aimn = -aim;   // no neg_a modifier on float WMMA
#pragma unroll
    for (int j = 0; j < 2; ++j) {
      const int nrow = n0 + j * 16 + (lane & 15);
      v16h bre = *(const v16h*)&sBre[cur][nrow][kh];
      v16h bim = *(const v16h*)&sBim[cur][nrow][kh];
      v16h bw  = *(const v16h*)&sW[cur][nrow][kh];
      // Re = psi_re·p_re + psi_im·p_im
      aRe[j] = __builtin_amdgcn_wmma_f32_16x16x32_f16(false, are,  false, bre,
                                                      (short)0, aRe[j], false, false);
      aRe[j] = __builtin_amdgcn_wmma_f32_16x16x32_f16(false, aim,  false, bim,
                                                      (short)0, aRe[j], false, false);
      // Im = psi_re·p_im - psi_im·p_re
      aIm[j] = __builtin_amdgcn_wmma_f32_16x16x32_f16(false, are,  false, bim,
                                                      (short)0, aIm[j], false, false);
      aIm[j] = __builtin_amdgcn_wmma_f32_16x16x32_f16(false, aimn, false, bre,
                                                      (short)0, aIm[j], false, false);
      // Linear = psi_re·dec_w
      aLn[j] = __builtin_amdgcn_wmma_f32_16x16x32_f16(false, are,  false, bw,
                                                      (short)0, aLn[j], false, false);
    }
    __syncthreads();
  }

#pragma unroll
  for (int j = 0; j < 2; ++j) {
#pragma unroll
    for (int r = 0; r < 8; ++r) {
      int srow = sbase + m0 + r + ((lane >> 4) << 3);
      int vcol = vbase + n0 + j * 16 + (lane & 15);
      float re = aRe[j][r], im = aIm[j][r];
      out[(size_t)srow * VOCAB + vcol] = re * re + im * im + aLn[j][r] + dec_b[vcol];
    }
  }
}

// ---------------------------------------------------------------------------
// Launch
// ---------------------------------------------------------------------------
extern "C" void kernel_launch(void* const* d_in, const int* in_sizes, int n_in,
                              void* d_out, int out_size, void* d_ws, size_t ws_size,
                              hipStream_t stream) {
  const int*   codes    = (const int*)d_in[0];
  const float* H        = (const float*)d_in[1];   // complex64 interleaved
  const float* hbar     = (const float*)d_in[2];
  const float* patterns = (const float*)d_in[3];   // complex64 interleaved
  const float* dec_w    = (const float*)d_in[4];
  const float* dec_b    = (const float*)d_in[5];
  (void)in_sizes; (void)n_in; (void)out_size; (void)ws_size;
  float* out = (float*)d_out;

  const size_t SD = (size_t)SEQ * DIM;
  float* pA_re = (float*)d_ws;            // ping
  float* pA_im = pA_re + SD;
  float* pB_re = pA_im + SD;              // pong
  float* pB_im = pB_re + SD;
  _Float16* p16_re = (_Float16*)(pB_im + SD);
  _Float16* p16_im = p16_re + SD;         // total ws: 64MB fp32 + 16MB f16 = 80MB

  // 1) wave init -> pA
  wave_init_kernel<<<(SEQ * (DIM / 4)) / 256, 256, 0, stream>>>(codes, pA_re, pA_im);

  // 2) 10 evolution steps: pA --evolve--> pB --normalize--> pA
  dim3 egrid(DIM / 64, SEQ / 64);
  for (int t = 0; t < 10; ++t) {
    evolve_kernel<<<egrid, 256, 0, stream>>>(pA_re, pA_im, H, hbar, pB_re, pB_im);
    normalize_kernel<<<SEQ, 256, 0, stream>>>(pB_re, pB_im, pA_re, pA_im);
  }

  // 3) convert to f16 for the decode GEMM
  cvt_kernel<<<(int)((SD + 255) / 256), 256, 0, stream>>>(pA_re, pA_im, p16_re, p16_im,
                                                          (int)SD);

  // 4) fused decode GEMM (x = S tiles fast, y = V tiles -> patterns read once)
  dim3 dgrid(SEQ / 64, VOCAB / 64);
  decode_kernel<<<dgrid, 256, 0, stream>>>(p16_re, p16_im, patterns, dec_w, dec_b, out);
}
